// MultiHeadAttention_257698038339
// MI455X (gfx1250) — matrix operand order repaired and checked by reading, not compiled, not run on hardware
//
#include <hip/hip_runtime.h>

// ---------------------------------------------------------------------------
// MHA for MI455X (gfx1250): bf16 WMMA pipeline, register-blocked GEMMs,
// double-buffered async-LDS flash attention.
//   S=2048, B=4, D=1024, H=16, d_k=64, M = S*B = 8192
// ---------------------------------------------------------------------------

#define SEQ    2048
#define BATCH  4
#define DMODEL 1024
#define NHEAD  16
#define DK     64
#define MROWS  (SEQ * BATCH)   // 8192

typedef __attribute__((ext_vector_type(16))) __bf16 bfx16;
typedef __attribute__((ext_vector_type(8)))  float  v8f;

union Frag16 {
    bfx16 v;
    uint4 q[2];
    unsigned int u[8];
};

__device__ __forceinline__ v8f wmma_bf16(const bfx16& a, const bfx16& b, v8f c) {
    return __builtin_amdgcn_wmma_f32_16x16x32_bf16(
        false, a, false, b, (short)0, c, false, false);
}

// A fragment (16x32 bf16): lane holds row M=lane%16.
// lanes 0-15: K = kk+0..15; lanes 16-31: K = kk+16..31
__device__ __forceinline__ Frag16 load_afrag(const __bf16* rowptr, int kk, int hi) {
    Frag16 f;
    const uint4* p = reinterpret_cast<const uint4*>(rowptr + kk + 16 * hi);
    f.q[0] = p[0];
    f.q[1] = p[1];
    return f;
}

// B fragment (32x16 bf16) from row-major (N x K) operand: lane holds col N=lane%16;
// lanes 0-15 cover K=kk..kk+15, lanes 16-31 cover kk+16..kk+31.
__device__ __forceinline__ Frag16 load_bfrag(const __bf16* rowptr, int kk, int hi) {
    Frag16 f;
    const uint4* p = reinterpret_cast<const uint4*>(rowptr + kk + 16 * hi);
    f.q[0] = p[0];
    f.q[1] = p[1];
    return f;
}

// ---------------------------------------------------------------------------
// gfx1250 async memory->LDS path (builtin verified present on this toolchain;
// signature: (int4 AS1* gsrc, int4 AS3* ldst, imm offset, imm cpol))
// ---------------------------------------------------------------------------
#if defined(__has_builtin)
# if __has_builtin(__builtin_amdgcn_global_load_async_to_lds_b128) && \
     __has_builtin(__builtin_amdgcn_s_wait_asynccnt)
#  define HAVE_ASYNC_LDS 1
# endif
#endif
#ifndef HAVE_ASYNC_LDS
# define HAVE_ASYNC_LDS 0
#endif

#if HAVE_ASYNC_LDS
typedef int v4i_t __attribute__((ext_vector_type(4)));
typedef __attribute__((address_space(1))) v4i_t gbl_v4i;
typedef __attribute__((address_space(3))) v4i_t lds_v4i;
#endif

// Stage 32 bytes (16 bf16) from global to LDS (2 async b128 ops per thread).
__device__ __forceinline__ void stage32(__bf16* l, const __bf16* g) {
#if HAVE_ASYNC_LDS
    __builtin_amdgcn_global_load_async_to_lds_b128(
        (gbl_v4i*)(v4i_t*)(__bf16*)g, (lds_v4i*)(v4i_t*)l, 0, 0);
    __builtin_amdgcn_global_load_async_to_lds_b128(
        (gbl_v4i*)(v4i_t*)((__bf16*)g + 8), (lds_v4i*)(v4i_t*)(l + 8), 0, 0);
#else
    reinterpret_cast<uint4*>(l)[0] = reinterpret_cast<const uint4*>(g)[0];
    reinterpret_cast<uint4*>(l)[1] = reinterpret_cast<const uint4*>(g)[1];
#endif
}

// ---------------------------------------------------------------------------
// Stage 0: f32 -> bf16 conversion (x4 vectorized)
// ---------------------------------------------------------------------------
__global__ __launch_bounds__(256) void cvt_f32_bf16_v4(const float4* __restrict__ in,
                                                       uint2* __restrict__ out, int n4) {
    int i = blockIdx.x * 256 + threadIdx.x;
    if (i < n4) {
        float4 f = in[i];
        union { __bf16 h[4]; uint2 u; } r;
        r.h[0] = (__bf16)f.x; r.h[1] = (__bf16)f.y;
        r.h[2] = (__bf16)f.z; r.h[3] = (__bf16)f.w;
        out[i] = r.u;
    }
}

// ---------------------------------------------------------------------------
// Stage 1: fused QKV projection, NT GEMM, 32x32 output super-tile per wave.
// Per K-chunk: 2 A frags + 6 B frags (16 b128 loads) -> 12 independent WMMAs.
// ---------------------------------------------------------------------------
__global__ __launch_bounds__(256) void qkv_kernel(
    const __bf16* __restrict__ xb,
    const __bf16* __restrict__ wq, const __bf16* __restrict__ wk,
    const __bf16* __restrict__ wv,
    const float* __restrict__ bq, const float* __restrict__ bk,
    const float* __restrict__ bv,
    __bf16* __restrict__ Qb, __bf16* __restrict__ Kb, __bf16* __restrict__ Vt) {
    const int lane = threadIdx.x & 31;
    const int wav  = threadIdx.x >> 5;
    const int gw   = blockIdx.x * 8 + wav;
    const int n2   = gw & 31;    // 32 super-tiles over N=1024 (32 cols)
    const int m2   = gw >> 5;    // 256 super-tiles over M=8192 (32 rows)
    const int li   = lane & 15;
    const int hi   = lane >> 4;

    const __bf16* arow[2];
    arow[0] = xb + (size_t)(m2 * 32 + li) * DMODEL;
    arow[1] = arow[0] + 16 * DMODEL;
    const __bf16* brow[3][2];
    brow[0][0] = wq + (size_t)(n2 * 32 + li) * DMODEL; brow[0][1] = brow[0][0] + 16 * DMODEL;
    brow[1][0] = wk + (size_t)(n2 * 32 + li) * DMODEL; brow[1][1] = brow[1][0] + 16 * DMODEL;
    brow[2][0] = wv + (size_t)(n2 * 32 + li) * DMODEL; brow[2][1] = brow[2][0] + 16 * DMODEL;

    v8f acc[3][2][2];
#pragma unroll
    for (int p = 0; p < 3; ++p)
#pragma unroll
        for (int mi = 0; mi < 2; ++mi)
#pragma unroll
            for (int ni = 0; ni < 2; ++ni) acc[p][mi][ni] = v8f{};

    for (int kk = 0; kk < DMODEL; kk += 32) {
        Frag16 a[2], b[3][2];
#pragma unroll
        for (int mi = 0; mi < 2; ++mi) a[mi] = load_afrag(arow[mi], kk, hi);
#pragma unroll
        for (int p = 0; p < 3; ++p)
#pragma unroll
            for (int ni = 0; ni < 2; ++ni) b[p][ni] = load_bfrag(brow[p][ni], kk, hi);
#pragma unroll
        for (int p = 0; p < 3; ++p)
#pragma unroll
            for (int mi = 0; mi < 2; ++mi)
#pragma unroll
                for (int ni = 0; ni < 2; ++ni)
                    acc[p][mi][ni] = wmma_bf16(a[mi].v, b[p][ni].v, acc[p][mi][ni]);
    }

#pragma unroll
    for (int ni = 0; ni < 2; ++ni) {
        const int n  = n2 * 32 + ni * 16 + li;
        const float vq = bq[n], vk = bk[n], vv = bv[n];
        const int h  = n >> 6;
        const int dd = n & 63;
#pragma unroll
        for (int mi = 0; mi < 2; ++mi) {
#pragma unroll
            for (int r = 0; r < 8; ++r) {
                const int m = m2 * 32 + mi * 16 + r + 8 * hi;  // m = s*BATCH + b
                const int s = m >> 2;
                const int b = m & 3;
                const size_t bhp = (size_t)(b * NHEAD + h);
                const float q = (acc[0][mi][ni][r] + vq) * 0.125f;  // 1/sqrt(64)
                const float k = acc[1][mi][ni][r] + vk;
                const float v = acc[2][mi][ni][r] + vv;
                Qb[(bhp * SEQ + s) * DK + dd] = (__bf16)q;
                Kb[(bhp * SEQ + s) * DK + dd] = (__bf16)k;
                Vt[(bhp * DK + dd) * SEQ + s] = (__bf16)v;
            }
        }
    }
}

// ---------------------------------------------------------------------------
// Stage 2: flash attention.
// Block = 4 waves, one (b,h), 128 q rows (each wave owns two 16-row q tiles).
// K (32x64) and V^T (64x32) tiles are DOUBLE-BUFFERED in LDS and filled with
// async-to-LDS DMA: chunk j+1's loads are issued before computing chunk j,
// then `s_wait_asynccnt 4` (4 async ops per thread per chunk, in-order
// completion) guarantees chunk j is resident while j+1 streams in.
// ---------------------------------------------------------------------------
__global__ __launch_bounds__(128) void attn_kernel(
    const __bf16* __restrict__ Qb, const __bf16* __restrict__ Kb,
    const __bf16* __restrict__ Vt, __bf16* __restrict__ Ob) {
    __shared__ __align__(16) __bf16 klds[2][32 * 72];   // 32 keys x 64 d, pad->72
    __shared__ __align__(16) __bf16 vlds[2][64 * 40];   // 64 d x 32 keys, pad->40
    __shared__ __align__(16) __bf16 plds[4][16 * 34];   // per-wave prob transpose

    const int tid  = threadIdx.x;
    const int lane = tid & 31;
    const int wav  = tid >> 5;
    const int li   = lane & 15;
    const int hi   = lane >> 4;
    const int bh   = blockIdx.x >> 4;   // 64 (b,h) pairs
    const int qg   = blockIdx.x & 15;   // 16 q groups of 128 rows

    const __bf16* qbase = Qb + (size_t)bh * SEQ * DK;
    const __bf16* kbase = Kb + (size_t)bh * SEQ * DK;
    const __bf16* vbase = Vt + (size_t)bh * DK * SEQ;

    const int s0 = (qg * 8 + wav * 2) * 16;  // rows s0 .. s0+31 (two q tiles)

    Frag16 qa[2][2];
#pragma unroll
    for (int t = 0; t < 2; ++t) {
        const __bf16* qrow = qbase + (size_t)(s0 + 16 * t + li) * DK;
        qa[t][0] = load_afrag(qrow, 0, hi);
        qa[t][1] = load_afrag(qrow, 32, hi);
    }

    v8f o[2][4];
    float mrow[2][8], lrow[2][8];
#pragma unroll
    for (int qi = 0; qi < 2; ++qi) {
#pragma unroll
        for (int n2 = 0; n2 < 4; ++n2) o[qi][n2] = v8f{};
#pragma unroll
        for (int r = 0; r < 8; ++r) { mrow[qi][r] = -1e30f; lrow[qi][r] = 0.0f; }
    }

    __bf16* pw = &plds[wav][0];

    // staging assignments (fixed per thread): 4 async b128 ops per chunk
    const int krow = tid >> 2, kcb = (tid & 3) * 16;    // K: 32 rows x 4 chunks
    const int vrow = tid >> 1, vcb = (tid & 1) * 16;    // V: 64 rows x 2 chunks

    // prologue: stage chunk 0 into buffer 0
    stage32(&klds[0][krow * 72 + kcb], kbase + (size_t)krow * DK + kcb);
    stage32(&vlds[0][vrow * 40 + vcb], vbase + (size_t)vrow * SEQ + vcb);

    for (int jj = 0; jj < SEQ / 32; ++jj) {
        const int cur = jj & 1;

        if (jj + 1 < SEQ / 32) {
            // issue next chunk's DMA into the other buffer, then wait for the
            // current chunk only (leave the 4 new ops in flight).
            const int j1 = (jj + 1) * 32;
            stage32(&klds[cur ^ 1][krow * 72 + kcb],
                    kbase + (size_t)(j1 + krow) * DK + kcb);
            stage32(&vlds[cur ^ 1][vrow * 40 + vcb],
                    vbase + (size_t)vrow * SEQ + j1 + vcb);
#if HAVE_ASYNC_LDS
            __builtin_amdgcn_s_wait_asynccnt(4);
#endif
        } else {
#if HAVE_ASYNC_LDS
            __builtin_amdgcn_s_wait_asynccnt(0);
#endif
        }
        __syncthreads();  // current chunk visible to all waves

        // K fragments (2 key tiles x 2 d-chunks) and V fragments, shared by
        // both q tiles of this wave.
        Frag16 kf[2][2];
#pragma unroll
        for (int t = 0; t < 2; ++t)
#pragma unroll
            for (int c = 0; c < 2; ++c) {
                const uint4* p = reinterpret_cast<const uint4*>(
                    &klds[cur][(16 * t + li) * 72 + c * 32 + 16 * hi]);
                kf[t][c].q[0] = p[0];
                kf[t][c].q[1] = p[1];
            }
        Frag16 vf[4];
#pragma unroll
        for (int n2 = 0; n2 < 4; ++n2) {
            const uint4* p = reinterpret_cast<const uint4*>(
                &vlds[cur][(n2 * 16 + li) * 40 + 16 * hi]);
            vf[n2].q[0] = p[0];
            vf[n2].q[1] = p[1];
        }

#pragma unroll
        for (int qi = 0; qi < 2; ++qi) {
            // ---- scores: two 16x16 tiles over d=64 ----
            v8f sc[2];
#pragma unroll
            for (int t = 0; t < 2; ++t) {
                v8f c = {};
                c = wmma_bf16(qa[qi][0].v, kf[t][0].v, c);
                c = wmma_bf16(qa[qi][1].v, kf[t][1].v, c);
                sc[t] = c;
            }

            // ---- online softmax over these 32 key columns ----
            float alpha[8];
#pragma unroll
            for (int r = 0; r < 8; ++r) {
                float mx = fmaxf(sc[0][r], sc[1][r]);
                mx = fmaxf(mx, __shfl_xor(mx, 1, 32));
                mx = fmaxf(mx, __shfl_xor(mx, 2, 32));
                mx = fmaxf(mx, __shfl_xor(mx, 4, 32));
                mx = fmaxf(mx, __shfl_xor(mx, 8, 32));
                const float nm = fmaxf(mrow[qi][r], mx);
                alpha[r]    = __expf(mrow[qi][r] - nm);
                mrow[qi][r] = nm;
            }
#pragma unroll
            for (int r = 0; r < 8; ++r) {
                const float p0 = __expf(sc[0][r] - mrow[qi][r]);
                const float p1 = __expf(sc[1][r] - mrow[qi][r]);
                float rs = p0 + p1;
                rs += __shfl_xor(rs, 1, 32);
                rs += __shfl_xor(rs, 2, 32);
                rs += __shfl_xor(rs, 4, 32);
                rs += __shfl_xor(rs, 8, 32);
                lrow[qi][r] = lrow[qi][r] * alpha[r] + rs;
                const int row = r + 8 * hi;
                pw[row * 34 + li]      = (__bf16)p0;
                pw[row * 34 + 16 + li] = (__bf16)p1;
                o[qi][0][r] *= alpha[r];
                o[qi][1][r] *= alpha[r];
                o[qi][2][r] *= alpha[r];
                o[qi][3][r] *= alpha[r];
            }
            __builtin_amdgcn_wave_barrier();  // same-wave DS ops are in-order

            // ---- gather P as 16x32 bf16 A fragment ----
            Frag16 pa;
#pragma unroll
            for (int v = 0; v < 8; ++v) {
                const int k = 2 * v + 16 * hi;
                pa.u[v] = *reinterpret_cast<const unsigned int*>(pw + li * 34 + k);
            }
            __builtin_amdgcn_wave_barrier();

            // ---- O += P * V ----
#pragma unroll
            for (int n2 = 0; n2 < 4; ++n2)
                o[qi][n2] = wmma_bf16(pa.v, vf[n2].v, o[qi][n2]);
        }

        __syncthreads();  // all waves done with buffer `cur` before re-staging
    }

    // ---- normalize and store to Ob (M x 1024, heads concatenated) ----
    const int b = bh >> 4;
    const int h = bh & 15;
#pragma unroll
    for (int qi = 0; qi < 2; ++qi) {
        float invl[8];
#pragma unroll
        for (int r = 0; r < 8; ++r) invl[r] = 1.0f / lrow[qi][r];
#pragma unroll
        for (int n2 = 0; n2 < 4; ++n2) {
            const int col = h * DK + n2 * 16 + li;
#pragma unroll
            for (int r = 0; r < 8; ++r) {
                const int srow = s0 + qi * 16 + r + 8 * hi;
                const int m    = srow * BATCH + b;
                Ob[(size_t)m * DMODEL + col] = (__bf16)(o[qi][n2][r] * invl[r]);
            }
        }
    }
}

// ---------------------------------------------------------------------------
// Stage 3: output projection, 64x64 output super-tile per wave.
// Per K-chunk: 8 fragments (16 b128 loads) -> 16 independent WMMAs.
// ---------------------------------------------------------------------------
__global__ __launch_bounds__(256) void oproj_kernel(
    const __bf16* __restrict__ Ob, const __bf16* __restrict__ wo,
    const float* __restrict__ bo, float* __restrict__ out) {
    const int lane = threadIdx.x & 31;
    const int wav  = threadIdx.x >> 5;
    const int gw   = blockIdx.x * 8 + wav;
    const int nt4  = gw & 15;    // 16 super-tiles over N=1024 (64 cols)
    const int mt4  = gw >> 4;    // 128 super-tiles over M=8192 (64 rows)
    const int li   = lane & 15;
    const int hi   = lane >> 4;

    const __bf16* arow[4];
    const __bf16* brow[4];
#pragma unroll
    for (int i = 0; i < 4; ++i) {
        arow[i] = Ob + (size_t)(mt4 * 64 + i * 16 + li) * DMODEL;
        brow[i] = wo + (size_t)(nt4 * 64 + i * 16 + li) * DMODEL;
    }

    v8f acc[4][4];
#pragma unroll
    for (int i = 0; i < 4; ++i)
#pragma unroll
        for (int j = 0; j < 4; ++j) acc[i][j] = v8f{};

    for (int kk = 0; kk < DMODEL; kk += 32) {
        Frag16 a[4], b[4];
#pragma unroll
        for (int i = 0; i < 4; ++i) a[i] = load_afrag(arow[i], kk, hi);
#pragma unroll
        for (int j = 0; j < 4; ++j) b[j] = load_bfrag(brow[j], kk, hi);
#pragma unroll
        for (int i = 0; i < 4; ++i)
#pragma unroll
            for (int j = 0; j < 4; ++j)
                acc[i][j] = wmma_bf16(a[i].v, b[j].v, acc[i][j]);
    }

#pragma unroll
    for (int j = 0; j < 4; ++j) {
        const int   n    = nt4 * 64 + j * 16 + li;
        const float bias = bo[n];
#pragma unroll
        for (int i = 0; i < 4; ++i) {
#pragma unroll
            for (int r = 0; r < 8; ++r) {
                const int m = mt4 * 64 + i * 16 + r + 8 * hi;
                out[(size_t)m * DMODEL + n] = acc[i][j][r] + bias;
            }
        }
    }
}

// ---------------------------------------------------------------------------
// Launch
// ---------------------------------------------------------------------------
extern "C" void kernel_launch(void* const* d_in, const int* in_sizes, int n_in,
                              void* d_out, int out_size, void* d_ws, size_t ws_size,
                              hipStream_t stream) {
    const float* x  = (const float*)d_in[0];
    const float* Wq = (const float*)d_in[1];
    const float* bq = (const float*)d_in[2];
    const float* Wk = (const float*)d_in[3];
    const float* bk = (const float*)d_in[4];
    const float* Wv = (const float*)d_in[5];
    const float* bv = (const float*)d_in[6];
    const float* Wo = (const float*)d_in[7];
    const float* bo = (const float*)d_in[8];
    float* out = (float*)d_out;

    char* ws = (char*)d_ws;
    const size_t MB = 1u << 20;
    __bf16* xb  = (__bf16*)(ws + 0 * MB);    // 16 MB
    __bf16* wqb = (__bf16*)(ws + 16 * MB);   //  2 MB
    __bf16* wkb = (__bf16*)(ws + 18 * MB);   //  2 MB
    __bf16* wvb = (__bf16*)(ws + 20 * MB);   //  2 MB
    __bf16* wob = (__bf16*)(ws + 22 * MB);   //  2 MB
    __bf16* Qb  = (__bf16*)(ws + 24 * MB);   // 16 MB (B,H,S,DK)
    __bf16* Kb  = (__bf16*)(ws + 40 * MB);   // 16 MB (B,H,S,DK)
    __bf16* Vt  = (__bf16*)(ws + 56 * MB);   // 16 MB (B,H,DK,S)
    __bf16* Ob  = (__bf16*)(ws + 72 * MB);   // 16 MB (M,D)

    const int NX4 = MROWS * DMODEL / 4;      // 2097152
    const int NW4 = DMODEL * DMODEL / 4;     //  262144

    cvt_f32_bf16_v4<<<NX4 / 256, 256, 0, stream>>>((const float4*)x, (uint2*)xb, NX4);
    cvt_f32_bf16_v4<<<NW4 / 256, 256, 0, stream>>>((const float4*)Wq, (uint2*)wqb, NW4);
    cvt_f32_bf16_v4<<<NW4 / 256, 256, 0, stream>>>((const float4*)Wk, (uint2*)wkb, NW4);
    cvt_f32_bf16_v4<<<NW4 / 256, 256, 0, stream>>>((const float4*)Wv, (uint2*)wvb, NW4);
    cvt_f32_bf16_v4<<<NW4 / 256, 256, 0, stream>>>((const float4*)Wo, (uint2*)wob, NW4);

    // 256 * 32 super-tiles / 8 waves per block
    qkv_kernel<<<1024, 256, 0, stream>>>(xb, wqb, wkb, wvb, bq, bk, bv, Qb, Kb, Vt);

    // 64 (b,h) * 16 q-groups (128 rows each)
    attn_kernel<<<1024, 128, 0, stream>>>(Qb, Kb, Vt, Ob);

    // 128 * 16 super-tiles / 8 waves per block
    oproj_kernel<<<256, 256, 0, stream>>>(Ob, wob, bo, out);
}
